// DualG_45561013076175
// MI455X (gfx1250) — matrix operand order repaired and checked by reading, not compiled, not run on hardware
//
#include <hip/hip_runtime.h>
#include <hip/hip_bf16.h>
#include <stdint.h>

#define NNODES 50000
#define NEDGES 250000
#define DFEAT  300
#define KPAD   320          // padded K for D=300  -> 10 k-tiles of 32
#define KT_D   10
#define KPAD2  608          // padded K for 2D=600 -> 19 k-tiles of 32
#define KT_2D  19
#define NTILES 20           // 320 padded output cols (300 real), 16 each
#define MTILES (NNODES / 16) // 3125 exact

typedef __attribute__((ext_vector_type(16))) __bf16 bf16x16;
typedef __attribute__((ext_vector_type(8)))  __bf16 bf16x8;
typedef __attribute__((ext_vector_type(8)))  float  f32x8;

// ---------- helpers ----------
__device__ inline bf16x16 load_pair(const __bf16* p0, const __bf16* p1) {
  bf16x8 lo = *(const bf16x8*)p0;
  bf16x8 hi = *(const bf16x8*)p1;
  bf16x16 r;
#pragma unroll
  for (int i = 0; i < 8; ++i) { r[i] = lo[i]; r[i + 8] = hi[i]; }
  return r;
}

// ---------- small utility kernels ----------
__global__ void zero_f32_kernel(float* p, int n) {
  int i = blockIdx.x * blockDim.x + threadIdx.x;
  if (i < n) p[i] = 0.f;
}

__global__ void deg_count_kernel(const int* __restrict__ src, const int* __restrict__ dst,
                                 float* dout, float* din) {
  int e = blockIdx.x * blockDim.x + threadIdx.x;
  if (e >= NEDGES) return;
  atomicAdd(&dout[src[e]], 1.f);
  atomicAdd(&din[dst[e]], 1.f);
}

__global__ void deg_final_kernel(float* d, int n) {
  int i = blockIdx.x * blockDim.x + threadIdx.x;
  if (i >= n) return;
  d[i] = rsqrtf(fmaxf(d[i], 1.f));
}

// h[i,k] = x[i,k] * dinv[i]
__global__ void scale_rows_kernel(const float* __restrict__ x, const float* __restrict__ dinv,
                                  float* __restrict__ h) {
  int t = blockIdx.x * blockDim.x + threadIdx.x;
  if (t >= NNODES * DFEAT) return;
  int i = t / DFEAT;
  h[t] = x[t] * dinv[i];
}

// agg[dst[e], k] += h[src[e], k]   (segment_sum over edges)
__global__ void scatter_add_kernel(const int* __restrict__ src, const int* __restrict__ dst,
                                   const float* __restrict__ h, float* __restrict__ agg) {
  int t = blockIdx.x * blockDim.x + threadIdx.x;
  if (t >= NEDGES * DFEAT) return;
  int e = t / DFEAT;
  int k = t - e * DFEAT;
  atomicAdd(&agg[(size_t)dst[e] * DFEAT + k], h[(size_t)src[e] * DFEAT + k]);
}

// Abf[i,k] = bf16(agg[i,k] * dinv[i]) for k<300, else 0  (row-major, stride KPAD)
__global__ void make_bf_in_kernel(const float* __restrict__ agg, const float* __restrict__ dinv,
                                  __bf16* __restrict__ out) {
  int t = blockIdx.x * blockDim.x + threadIdx.x;
  if (t >= NNODES * KPAD) return;
  int i = t / KPAD;
  int k = t - i * KPAD;
  float v = (k < DFEAT) ? agg[(size_t)i * DFEAT + k] * dinv[i] : 0.f;
  out[t] = (__bf16)v;
}

// zero bf16 padding cols 600..607 of the concat buffer
__global__ void cc_tail_kernel(__bf16* cc) {
  int t = blockIdx.x * blockDim.x + threadIdx.x;
  if (t >= NNODES * 8) return;
  int i = t >> 3, j = t & 7;
  cc[(size_t)i * KPAD2 + 600 + j] = (__bf16)0.f;
}

// Pack weight W[K x 300] (row-major) into WMMA-B fragment order:
// idx = ((kt*NTILES + nt)*32 + lane)*16 + j ; element (k = kt*32 + (lane>=16?16:0) + j,
//                                             n = nt*16 + (lane&15)), zero padded.
__global__ void pack_w_kernel(const float* __restrict__ W, int K, int ktiles,
                              __bf16* __restrict__ Bp) {
  int t = blockIdx.x * blockDim.x + threadIdx.x;
  int total = ktiles * NTILES * 512;
  if (t >= total) return;
  int j = t & 15;
  int lane = (t >> 4) & 31;
  int nt = (t >> 9) % NTILES;
  int kt = (t >> 9) / NTILES;
  int k = kt * 32 + ((lane >= 16) ? 16 : 0) + j;
  int n = nt * 16 + (lane & 15);
  float v = (k < K && n < DFEAT) ? W[(size_t)k * DFEAT + n] : 0.f;
  Bp[t] = (__bf16)v;
}

// ---------- WMMA GEMM: out = act(A[N x lda]bf16 @ W + bias) ----------
// one wave -> 16 rows x 64 cols; 4 waves/block; grid = (ceil(MTILES/4), NTILES/4)
__device__ inline void epilogue_tile(const f32x8& acc, int nt, int rbase, int colb,
                                     const float* __restrict__ bias,
                                     float* __restrict__ outF, __bf16* __restrict__ outBf,
                                     int relu) {
  int col = nt * 16 + colb;
  float bv = (col < DFEAT) ? bias[col] : 0.f;
#pragma unroll
  for (int r = 0; r < 8; ++r) {
    float v = acc[r] + bv;
    if (relu) v = fmaxf(v, 0.f);
    int row = rbase + r;
    if (outF && col < DFEAT) outF[(size_t)row * DFEAT + col] = v;
    if (outBf) outBf[(size_t)row * KPAD + col] = (__bf16)((col < DFEAT) ? v : 0.f);
  }
}

__launch_bounds__(128)
__global__ void gemm_bf16_kernel(const __bf16* __restrict__ A, int lda,
                                 const __bf16* __restrict__ Bp, int ktiles,
                                 const float* __restrict__ bias,
                                 float* __restrict__ outF, __bf16* __restrict__ outBf,
                                 int relu) {
  const int lane = threadIdx.x & 31;
  const int wave = threadIdx.x >> 5;
  const int mtile = blockIdx.x * 4 + wave;
  if (mtile >= MTILES) return;                  // uniform per-wave exit; EXEC stays full
  const int nt0 = blockIdx.y * 4;
  const int m0 = mtile * 16;
  const int kb = (lane >> 4) * 16;              // A-frag K sub-offset per half-wave
  const __bf16* arow = A + (size_t)(m0 + (lane & 15)) * lda;

  f32x8 acc0 = {}, acc1 = {}, acc2 = {}, acc3 = {};
  for (int kt = 0; kt < ktiles; ++kt) {
    const int k0 = kt * 32;
    bf16x16 a = load_pair(arow + k0 + kb, arow + k0 + kb + 8);
    const __bf16* bb = Bp + (((size_t)kt * NTILES + nt0) * 32 + lane) * 16;
    bf16x16 b0 = load_pair(bb,        bb + 8);
    bf16x16 b1 = load_pair(bb +  512, bb +  520);
    bf16x16 b2 = load_pair(bb + 1024, bb + 1032);
    bf16x16 b3 = load_pair(bb + 1536, bb + 1544);
    acc0 = __builtin_amdgcn_wmma_f32_16x16x32_bf16(false, a, false, b0, (short)0, acc0, false, false);
    acc1 = __builtin_amdgcn_wmma_f32_16x16x32_bf16(false, a, false, b1, (short)0, acc1, false, false);
    acc2 = __builtin_amdgcn_wmma_f32_16x16x32_bf16(false, a, false, b2, (short)0, acc2, false, false);
    acc3 = __builtin_amdgcn_wmma_f32_16x16x32_bf16(false, a, false, b3, (short)0, acc3, false, false);
  }
  const int rbase = m0 + ((lane >> 4) << 3);    // C layout: lanes 16..31 hold M=8..15
  const int colb = lane & 15;
  epilogue_tile(acc0, nt0 + 0, rbase, colb, bias, outF, outBf, relu);
  epilogue_tile(acc1, nt0 + 1, rbase, colb, bias, outF, outBf, relu);
  epilogue_tile(acc2, nt0 + 2, rbase, colb, bias, outF, outBf, relu);
  epilogue_tile(acc3, nt0 + 3, rbase, colb, bias, outF, outBf, relu);
}

// ---------- LayerNorm: wave-per-row ----------
__global__ void ln_kernel(const float* __restrict__ X,
                          const float* __restrict__ gam, const float* __restrict__ bet,
                          float* __restrict__ outF,
                          __bf16* __restrict__ outBf, int ldbf, int coloff) {
  int lane = threadIdx.x & 31;
  int row = blockIdx.x * 8 + (threadIdx.x >> 5);
  if (row >= NNODES) return;
  const float* x = X + (size_t)row * DFEAT;
  float s = 0.f;
  for (int k = lane; k < DFEAT; k += 32) s += x[k];
  for (int o = 16; o; o >>= 1) s += __shfl_xor(s, o, 32);
  float mu = s * (1.0f / DFEAT);
  float vs = 0.f;
  for (int k = lane; k < DFEAT; k += 32) { float t = x[k] - mu; vs += t * t; }
  for (int o = 16; o; o >>= 1) vs += __shfl_xor(vs, o, 32);
  float rinv = rsqrtf(vs * (1.0f / DFEAT) + 1e-5f);
  for (int k = lane; k < DFEAT; k += 32) {
    float y = (x[k] - mu) * rinv * gam[k] + bet[k];
    if (outF) outF[(size_t)row * DFEAT + k] = y;
    if (outBf) outBf[(size_t)row * ldbf + coloff + k] = (__bf16)y;
  }
}

// ---------- AdaptiveFusion: wave-per-row ----------
__global__ void fusion_kernel(const float* __restrict__ r2, const float* __restrict__ c2,
                              const float* __restrict__ w1, const float* __restrict__ b1,
                              const float* __restrict__ w2, const float* __restrict__ b2,
                              float* __restrict__ out) {
  int lane = threadIdx.x & 31;
  int row = blockIdx.x * 8 + (threadIdx.x >> 5);
  if (row >= NNODES) return;
  const float* a = r2 + (size_t)row * DFEAT;
  const float* c = c2 + (size_t)row * DFEAT;
  float sa = 0.f, sb = 0.f;
  for (int k = lane; k < DFEAT; k += 32) { sa += a[k] * w1[k]; sb += c[k] * w2[k]; }
  for (int o = 16; o; o >>= 1) { sa += __shfl_xor(sa, o, 32); sb += __shfl_xor(sb, o, 32); }
  float alpha = 1.f / (1.f + __expf(-(sa + b1[0])));
  float beta  = 1.f / (1.f + __expf(-(sb + b2[0])));
  float al = alpha / (alpha + beta);
  for (int k = lane; k < DFEAT; k += 32)
    out[(size_t)row * DFEAT + k] = al * a[k] + (1.f - al) * c[k];
}

// ---------- host ----------
#define CDIV(a, b) (((a) + (b) - 1) / (b))

extern "C" void kernel_launch(void* const* d_in, const int* in_sizes, int n_in,
                              void* d_out, int out_size, void* d_ws, size_t ws_size,
                              hipStream_t stream) {
  (void)in_sizes; (void)n_in; (void)out_size; (void)ws_size;
  const float* table = (const float*)d_in[0];
  const int* row_src = (const int*)d_in[1];
  const int* row_dst = (const int*)d_in[2];
  const int* col_src = (const int*)d_in[3];
  const int* col_dst = (const int*)d_in[4];
  const float* W_row1 = (const float*)d_in[5];  const float* b_row1 = (const float*)d_in[6];
  const float* W_col1 = (const float*)d_in[7];  const float* b_col1 = (const float*)d_in[8];
  const float* rs_W = (const float*)d_in[9];    const float* rs_b = (const float*)d_in[10];
  const float* rs_g = (const float*)d_in[11];   const float* rs_bt = (const float*)d_in[12];
  const float* cs_W = (const float*)d_in[13];   const float* cs_b = (const float*)d_in[14];
  const float* cs_g = (const float*)d_in[15];   const float* cs_bt = (const float*)d_in[16];
  const float* gm_W = (const float*)d_in[17];   const float* gm_b = (const float*)d_in[18];
  const float* gm_g = (const float*)d_in[19];   const float* gm_bt = (const float*)d_in[20];
  const float* W_row2 = (const float*)d_in[21]; const float* b_row2 = (const float*)d_in[22];
  const float* W_col2 = (const float*)d_in[23]; const float* b_col2 = (const float*)d_in[24];
  const float* af_w1 = (const float*)d_in[25];  const float* af_b1 = (const float*)d_in[26];
  const float* af_w2 = (const float*)d_in[27];  const float* af_b2 = (const float*)d_in[28];

  // workspace carve-out
  char* base = (char*)d_ws;
  size_t off = 0;
  auto alloc = [&](size_t bytes) -> void* {
    void* p = base + off;
    off += (bytes + 255) & ~(size_t)255;
    return p;
  };
  float* dinvRout = (float*)alloc((size_t)NNODES * 4);
  float* dinvRin  = (float*)alloc((size_t)NNODES * 4);
  float* dinvCout = (float*)alloc((size_t)NNODES * 4);
  float* dinvCin  = (float*)alloc((size_t)NNODES * 4);
  const size_t packD = (size_t)KT_D * NTILES * 512 * 2;   // bytes
  __bf16* pk_row1 = (__bf16*)alloc(packD);
  __bf16* pk_col1 = (__bf16*)alloc(packD);
  __bf16* pk_rs   = (__bf16*)alloc(packD);
  __bf16* pk_cs   = (__bf16*)alloc(packD);
  __bf16* pk_row2 = (__bf16*)alloc(packD);
  __bf16* pk_col2 = (__bf16*)alloc(packD);
  __bf16* pk_gm   = (__bf16*)alloc((size_t)KT_2D * NTILES * 512 * 2);
  float*  h    = (float*)alloc((size_t)NNODES * DFEAT * 4);
  float*  agg  = (float*)alloc((size_t)NNODES * DFEAT * 4);   // also reused as pre-LN tmp
  __bf16* bfIn = (__bf16*)alloc((size_t)NNODES * KPAD * 2);
  __bf16* r1bf = (__bf16*)alloc((size_t)NNODES * KPAD * 2);
  __bf16* c1bf = (__bf16*)alloc((size_t)NNODES * KPAD * 2);
  __bf16* cc   = (__bf16*)alloc((size_t)NNODES * KPAD2 * 2);
  float*  g    = (float*)alloc((size_t)NNODES * DFEAT * 4);
  float*  c2   = (float*)alloc((size_t)NNODES * DFEAT * 4);
  float*  tmpF = agg;                 // aliased: pre-LN tmp, dead when agg is reused
  float*  r2   = (float*)d_out;       // r2 lives in d_out; fusion rewrites in place

  const int TPB = 256;
  const dim3 gGrid(CDIV(MTILES, 4), NTILES / 4);
  const dim3 gBlock(128);

  // ---- pack weights to WMMA-B fragment layout ----
  int packDthreads = KT_D * NTILES * 512;
  int packGthreads = KT_2D * NTILES * 512;
  pack_w_kernel<<<CDIV(packDthreads, TPB), TPB, 0, stream>>>(W_row1, DFEAT, KT_D, pk_row1);
  pack_w_kernel<<<CDIV(packDthreads, TPB), TPB, 0, stream>>>(W_col1, DFEAT, KT_D, pk_col1);
  pack_w_kernel<<<CDIV(packDthreads, TPB), TPB, 0, stream>>>(rs_W,   DFEAT, KT_D, pk_rs);
  pack_w_kernel<<<CDIV(packDthreads, TPB), TPB, 0, stream>>>(cs_W,   DFEAT, KT_D, pk_cs);
  pack_w_kernel<<<CDIV(packDthreads, TPB), TPB, 0, stream>>>(W_row2, DFEAT, KT_D, pk_row2);
  pack_w_kernel<<<CDIV(packDthreads, TPB), TPB, 0, stream>>>(W_col2, DFEAT, KT_D, pk_col2);
  pack_w_kernel<<<CDIV(packGthreads, TPB), TPB, 0, stream>>>(gm_W, 2 * DFEAT, KT_2D, pk_gm);

  // ---- degree normalizers ----
  zero_f32_kernel<<<CDIV(NNODES, TPB), TPB, 0, stream>>>(dinvRout, NNODES);
  zero_f32_kernel<<<CDIV(NNODES, TPB), TPB, 0, stream>>>(dinvRin, NNODES);
  zero_f32_kernel<<<CDIV(NNODES, TPB), TPB, 0, stream>>>(dinvCout, NNODES);
  zero_f32_kernel<<<CDIV(NNODES, TPB), TPB, 0, stream>>>(dinvCin, NNODES);
  deg_count_kernel<<<CDIV(NEDGES, TPB), TPB, 0, stream>>>(row_src, row_dst, dinvRout, dinvRin);
  deg_count_kernel<<<CDIV(NEDGES, TPB), TPB, 0, stream>>>(col_src, col_dst, dinvCout, dinvCin);
  deg_final_kernel<<<CDIV(NNODES, TPB), TPB, 0, stream>>>(dinvRout, NNODES);
  deg_final_kernel<<<CDIV(NNODES, TPB), TPB, 0, stream>>>(dinvRin, NNODES);
  deg_final_kernel<<<CDIV(NNODES, TPB), TPB, 0, stream>>>(dinvCout, NNODES);
  deg_final_kernel<<<CDIV(NNODES, TPB), TPB, 0, stream>>>(dinvCin, NNODES);

  const int ND = NNODES * DFEAT;
  const int ED = NEDGES * DFEAT;
  const int NK = NNODES * KPAD;

  auto graph_conv = [&](const float* x, const int* src, const int* dst,
                        const float* dinv_out, const float* dinv_in,
                        __bf16* pkW, const float* bias,
                        float* outF, __bf16* outBf) {
    scale_rows_kernel<<<CDIV(ND, TPB), TPB, 0, stream>>>(x, dinv_out, h);
    zero_f32_kernel<<<CDIV(ND, TPB), TPB, 0, stream>>>(agg, ND);
    scatter_add_kernel<<<CDIV(ED, TPB), TPB, 0, stream>>>(src, dst, h, agg);
    make_bf_in_kernel<<<CDIV(NK, TPB), TPB, 0, stream>>>(agg, dinv_in, bfIn);
    gemm_bf16_kernel<<<gGrid, gBlock, 0, stream>>>(bfIn, KPAD, pkW, KT_D, bias, outF, outBf, 1);
  };

  // ---- layer 1 ----
  graph_conv(table, row_src, row_dst, dinvRout, dinvRin, pk_row1, b_row1, nullptr, r1bf);
  graph_conv(table, col_src, col_dst, dinvCout, dinvCin, pk_col1, b_col1, nullptr, c1bf);

  // rs = LN(r1 @ rs_W + rs_b) -> cc[:, 0:300] (bf16)
  gemm_bf16_kernel<<<gGrid, gBlock, 0, stream>>>(r1bf, KPAD, pk_rs, KT_D, rs_b, tmpF, nullptr, 0);
  ln_kernel<<<CDIV(NNODES, 8), 256, 0, stream>>>(tmpF, rs_g, rs_bt, nullptr, cc, KPAD2, 0);
  // cs = LN(c1 @ cs_W + cs_b) -> cc[:, 300:600]
  gemm_bf16_kernel<<<gGrid, gBlock, 0, stream>>>(c1bf, KPAD, pk_cs, KT_D, cs_b, tmpF, nullptr, 0);
  ln_kernel<<<CDIV(NNODES, 8), 256, 0, stream>>>(tmpF, cs_g, cs_bt, nullptr, cc, KPAD2, 300);
  cc_tail_kernel<<<CDIV(NNODES * 8, TPB), TPB, 0, stream>>>(cc);

  // g = LN(cc @ gm_W + gm_b)
  gemm_bf16_kernel<<<gGrid, gBlock, 0, stream>>>(cc, KPAD2, pk_gm, KT_2D, gm_b, tmpF, nullptr, 0);
  ln_kernel<<<CDIV(NNODES, 8), 256, 0, stream>>>(tmpF, gm_g, gm_bt, g, nullptr, 0, 0);

  // ---- layer 2 ----
  graph_conv(g, row_src, row_dst, dinvRout, dinvRin, pk_row2, b_row2, r2, nullptr);
  graph_conv(g, col_src, col_dst, dinvCout, dinvCin, pk_col2, b_col2, c2, nullptr);

  // ---- adaptive fusion -> d_out (in place over r2) ----
  fusion_kernel<<<CDIV(NNODES, 8), 256, 0, stream>>>(r2, c2, af_w1, af_b1, af_w2, af_b2,
                                                     (float*)d_out);
}
